// Block_1511828488887
// MI455X (gfx1250) — matrix operand order repaired and checked by reading, not compiled, not run on hardware
//
#include <hip/hip_runtime.h>
#include <math.h>

typedef __attribute__((ext_vector_type(16))) __bf16 v16bf;
typedef __attribute__((ext_vector_type(8)))  __bf16 v8bf;
typedef __attribute__((ext_vector_type(8)))  float  v8f;

#define B_   4
#define H_   56
#define W_   56
#define N_   3136
#define C_   384
#define NH_  12
#define HD_  32
#define PL_  49
#define LL_  9
#define HID_ 1024
#define M_   (B_*N_)

__device__ __forceinline__ float gelu_exact(float x) {
  return 0.5f * x * (1.0f + erff(x * 0.7071067811865476f));
}

// ---------------------------------------------------------------------------
// LayerNorm: f32 [M,C] -> bf16 [M,C].  One block (128 threads) per row.
// ---------------------------------------------------------------------------
__global__ __launch_bounds__(128) void ln_bf16_kernel(
    const float* __restrict__ in, const float* __restrict__ g,
    const float* __restrict__ b, __bf16* __restrict__ out, int M, int C) {
  int row = blockIdx.x;
  if (row >= M) return;
  const float* r = in + (size_t)row * C;
  float s = 0.f, ss = 0.f;
  for (int i = threadIdx.x; i < C; i += 128) {
    float v = r[i];
    s += v; ss += v * v;
  }
  __shared__ float s1[128], s2[128];
  s1[threadIdx.x] = s; s2[threadIdx.x] = ss;
  __syncthreads();
  for (int off = 64; off > 0; off >>= 1) {
    if (threadIdx.x < off) {
      s1[threadIdx.x] += s1[threadIdx.x + off];
      s2[threadIdx.x] += s2[threadIdx.x + off];
    }
    __syncthreads();
  }
  float mean = s1[0] / C;
  float var = s2[0] / C - mean * mean;
  float rstd = rsqrtf(var + 1e-5f);
  __bf16* o = out + (size_t)row * C;
  for (int i = threadIdx.x; i < C; i += 128)
    o[i] = (__bf16)(((r[i] - mean) * rstd) * g[i] + b[i]);
}

// ---------------------------------------------------------------------------
// WMMA GEMM: out[M,Nn] = act(A[M,K](bf16) @ W[Nn,K]^T + bias) (+res)
// Block tile 128x64, BK=32. 8 waves, each owns a 32x32 output (4 WMMA accums).
// A tile staged with GLOBAL_LOAD_ASYNC_TO_LDS_B128 (ASYNCcnt, no VGPR
// round-trip); W tile staged via VGPRs for the f32->bf16 convert.
// LDS layout keeps every fragment as contiguous 16B runs -> ds_load_b128.
// ---------------------------------------------------------------------------
__global__ __launch_bounds__(256) void gemm_bf16_kernel(
    const __bf16* __restrict__ A, const float* __restrict__ W,
    const float* __restrict__ bias, const float* __restrict__ residual,
    float* __restrict__ outF, __bf16* __restrict__ outB,
    int M, int Nn, int K, int act) {
  __shared__ __bf16 As[128][32 + 8];  // row stride 80B (16B-aligned)
  __shared__ __bf16 Bs[64][32 + 8];   // [n][k]

  int tid = threadIdx.x;
  int wave = tid >> 5, lane = tid & 31;
  int wm = wave & 3, wn = wave >> 2;       // 4x2 wave grid
  int rowBlock = blockIdx.y * 128, colBlock = blockIdx.x * 64;
  int l16 = lane & 15, hs = lane >> 4;

  v8f c00 = {}, c01 = {}, c10 = {}, c11 = {};

  // A loader: 128 rows x 32 k, 32B per thread (two async b128 copies)
  int alr = tid >> 1;            // 0..127
  int alc = (tid & 1) * 16;      // 0,16
  // Rows never cross-contaminate outputs: clamp instead of zero-fill.
  int ar = rowBlock + alr;
  if (ar >= M) ar = M - 1;
  const __bf16* agsrc = A + (size_t)ar * K + alc;
  unsigned alds = (unsigned)(uintptr_t)&As[alr][alc];  // LDS byte offset
  // B loader: 64 rows x 32 k f32, 8 per thread
  int blr = tid >> 2;            // 0..63
  int blc = (tid & 3) * 8;       // 0,8,16,24

  for (int k0 = 0; k0 < K; k0 += 32) {
    // ---- stage A tile: async global -> LDS (offset applies to both sides)
    asm volatile(
        "global_load_async_to_lds_b128 %0, %1, off\n\t"
        "global_load_async_to_lds_b128 %0, %1, off offset:16"
        :
        : "v"(alds), "v"((unsigned long long)(uintptr_t)(agsrc + k0))
        : "memory");
    // ---- stage W tile: Bs[n][k] = W[colBlock+n][k0+k]  (f32 -> bf16)
    {
      const float* wsrc = W + (size_t)(colBlock + blr) * K + k0 + blc;
      v8f wf = *(const v8f*)wsrc;
      v8bf wb;
#pragma unroll
      for (int e = 0; e < 8; e++) wb[e] = (__bf16)wf[e];
      *(v8bf*)&Bs[blr][blc] = wb;
      if (k0 + 32 < K) __builtin_prefetch(wsrc + 32, 0, 3);
    }
    asm volatile("s_wait_asynccnt 0x0" ::: "memory");
    __syncthreads();

    // ---- fragments (ISA 7.12.2 layouts), contiguous 16B runs
    int am0 = wm * 32 + l16, am1 = am0 + 16;
    v8bf a0lo = *(const v8bf*)&As[am0][hs * 8];
    v8bf a0hi = *(const v8bf*)&As[am0][16 + hs * 8];
    v8bf a1lo = *(const v8bf*)&As[am1][hs * 8];
    v8bf a1hi = *(const v8bf*)&As[am1][16 + hs * 8];
    int bn0 = wn * 32 + l16, bn1 = bn0 + 16;
    v8bf b0lo = *(const v8bf*)&Bs[bn0][hs * 8];
    v8bf b0hi = *(const v8bf*)&Bs[bn0][16 + hs * 8];
    v8bf b1lo = *(const v8bf*)&Bs[bn1][hs * 8];
    v8bf b1hi = *(const v8bf*)&Bs[bn1][16 + hs * 8];
    v16bf a0 = __builtin_shufflevector(a0lo, a0hi, 0, 1, 2, 3, 4, 5, 6, 7, 8,
                                       9, 10, 11, 12, 13, 14, 15);
    v16bf a1 = __builtin_shufflevector(a1lo, a1hi, 0, 1, 2, 3, 4, 5, 6, 7, 8,
                                       9, 10, 11, 12, 13, 14, 15);
    v16bf b0 = __builtin_shufflevector(b0lo, b0hi, 0, 1, 2, 3, 4, 5, 6, 7, 8,
                                       9, 10, 11, 12, 13, 14, 15);
    v16bf b1 = __builtin_shufflevector(b1lo, b1hi, 0, 1, 2, 3, 4, 5, 6, 7, 8,
                                       9, 10, 11, 12, 13, 14, 15);

    c00 = __builtin_amdgcn_wmma_f32_16x16x32_bf16(false, a0, false, b0,
                                                  (short)0, c00, false, false);
    c01 = __builtin_amdgcn_wmma_f32_16x16x32_bf16(false, a0, false, b1,
                                                  (short)0, c01, false, false);
    c10 = __builtin_amdgcn_wmma_f32_16x16x32_bf16(false, a1, false, b0,
                                                  (short)0, c10, false, false);
    c11 = __builtin_amdgcn_wmma_f32_16x16x32_bf16(false, a1, false, b1,
                                                  (short)0, c11, false, false);
    __syncthreads();
  }

  // ---- epilogue: C/D layout — lane col = l16, VGPR i row = i + hs*8
  int col0 = colBlock + wn * 32 + l16;
  int col1 = col0 + 16;
  float bias0 = bias ? bias[col0] : 0.f;
  float bias1 = bias ? bias[col1] : 0.f;
#pragma unroll
  for (int fr = 0; fr < 2; fr++) {
    v8f cc0 = fr ? c10 : c00;
    v8f cc1 = fr ? c11 : c01;
#pragma unroll
    for (int i = 0; i < 8; i++) {
      int row = rowBlock + wm * 32 + fr * 16 + hs * 8 + i;
      if (row < M) {
        float v0 = cc0[i] + bias0;
        float v1 = cc1[i] + bias1;
        if (act == 1) { v0 = gelu_exact(v0); v1 = gelu_exact(v1); }
        if (residual) {
          v0 += residual[(size_t)row * Nn + col0];
          v1 += residual[(size_t)row * Nn + col1];
        }
        if (outF) {
          outF[(size_t)row * Nn + col0] = v0;
          outF[(size_t)row * Nn + col1] = v1;
        }
        if (outB) {
          outB[(size_t)row * Nn + col0] = (__bf16)v0;
          outB[(size_t)row * Nn + col1] = (__bf16)v1;
        }
      }
    }
  }
}

// ---------------------------------------------------------------------------
// q post: per (m,h) l2-normalize, add query embedding, scale by softplus*sls.
// ---------------------------------------------------------------------------
__global__ void q_post_kernel(float* __restrict__ q,
                              const float* __restrict__ temp,
                              const float* __restrict__ qe,
                              const float* __restrict__ sls) {
  int idx = blockIdx.x * blockDim.x + threadIdx.x;
  if (idx >= M_ * NH_) return;
  int m = idx / NH_, h = idx % NH_;
  float* row = q + (size_t)m * C_ + h * HD_;
  float ss = 0.f;
#pragma unroll
  for (int d = 0; d < HD_; d++) ss += row[d] * row[d];
  float inv = 1.0f / fmaxf(sqrtf(ss), 1e-12f);
  float scale = log1pf(expf(temp[h])) * sls[0];
#pragma unroll
  for (int d = 0; d < HD_; d++)
    row[d] = (row[d] * inv + qe[h * HD_ + d]) * scale;
}

// per-head l2 normalize the k half of a [rows, 2C] kv buffer, in place
__global__ void k_norm_kernel(float* __restrict__ kv, int rows) {
  int idx = blockIdx.x * blockDim.x + threadIdx.x;
  if (idx >= rows * NH_) return;
  int r = idx / NH_, h = idx % NH_;
  float* row = kv + (size_t)r * (2 * C_) + h * HD_;
  float ss = 0.f;
#pragma unroll
  for (int d = 0; d < HD_; d++) ss += row[d] * row[d];
  float inv = 1.0f / fmaxf(sqrtf(ss), 1e-12f);
#pragma unroll
  for (int d = 0; d < HD_; d++) row[d] *= inv;
}

// 8x8 mean pool: s[B,N,C] -> pooled[B,PL,C]
__global__ void pool_kernel(const float* __restrict__ s,
                            float* __restrict__ pooled) {
  int idx = blockIdx.x * blockDim.x + threadIdx.x;
  if (idx >= B_ * PL_ * C_) return;
  int b = idx / (PL_ * C_);
  int rem = idx % (PL_ * C_);
  int p = rem / C_, c = rem % C_;
  int py = p / (W_ / 8), px = p % (W_ / 8);
  float acc = 0.f;
  for (int iy = 0; iy < 8; iy++)
    for (int ix = 0; ix < 8; ix++) {
      int n = (py * 8 + iy) * W_ + (px * 8 + ix);
      acc += s[((size_t)b * N_ + n) * C_ + c];
    }
  pooled[((size_t)b * PL_ + p) * C_ + c] = acc * (1.0f / 64.0f);
}

// CPB MLP: cpb[t,h] = relu(rct[t]@w1^T + b1) @ w2^T + b2
__global__ void cpb_kernel(const float* __restrict__ rct,
                           const float* __restrict__ w1,
                           const float* __restrict__ b1,
                           const float* __restrict__ w2,
                           const float* __restrict__ b2,
                           float* __restrict__ cpb, int T) {
  int t = blockIdx.x * blockDim.x + threadIdx.x;
  if (t >= T) return;
  float x0 = rct[t * 2], x1 = rct[t * 2 + 1];
  float acc[NH_];
#pragma unroll
  for (int h = 0; h < NH_; h++) acc[h] = b2[h];
  for (int j = 0; j < 512; j++) {
    float hv = fmaxf(x0 * w1[j * 2] + x1 * w1[j * 2 + 1] + b1[j], 0.f);
#pragma unroll
    for (int h = 0; h < NH_; h++) acc[h] += hv * w2[h * 512 + j];
  }
#pragma unroll
  for (int h = 0; h < NH_; h++) cpb[(size_t)t * NH_ + h] = acc[h];
}

// ---------------------------------------------------------------------------
// Attention core: one thread per (b,h,n). 9 local + 49 pooled scores,
// softmax(58), learnable-token term, weighted V sums. Output bf16 [M,C].
// q_norm is reconstructed as q_s/scale - query_embedding.
// ---------------------------------------------------------------------------
__global__ void attn_kernel(const float* __restrict__ qs,
                            const float* __restrict__ kv,
                            const float* __restrict__ kvp,
                            const float* __restrict__ cpb,
                            const int* __restrict__ rpi,
                            const unsigned char* __restrict__ pmask,
                            const float* __restrict__ rpb,
                            const float* __restrict__ lt,
                            const float* __restrict__ lb,
                            const float* __restrict__ temp,
                            const float* __restrict__ sls,
                            const float* __restrict__ qe,
                            __bf16* __restrict__ obuf) {
  int idx = blockIdx.x * blockDim.x + threadIdx.x;
  if (idx >= B_ * NH_ * N_) return;
  int b = idx / (NH_ * N_);
  int rem = idx % (NH_ * N_);
  int h = rem / N_, n = rem % N_;
  int y0 = n / W_, x0 = n % W_;

  float q[HD_];
  const float* qrow = qs + ((size_t)b * N_ + n) * C_ + h * HD_;
#pragma unroll
  for (int d = 0; d < HD_; d++) q[d] = qrow[d];
  float scale = log1pf(expf(temp[h])) * sls[0];
  float invs = 1.0f / scale;

  int nbr[LL_];
  bool ok[LL_];
  float sc[LL_ + PL_];
#pragma unroll
  for (int l = 0; l < LL_; l++) {
    int dy = l / 3 - 1, dx = l % 3 - 1;
    int yy = y0 + dy, xx = x0 + dx;
    ok[l] = (yy >= 0 && yy < H_ && xx >= 0 && xx < W_);
    nbr[l] = ok[l] ? (yy * W_ + xx) : 0;
    float v = rpb[h * LL_ + l];
    if (ok[l]) {
      const float* k = kv + ((size_t)b * N_ + nbr[l]) * (2 * C_) + h * HD_;
      float acc = 0.f;
#pragma unroll
      for (int d = 0; d < HD_; d++) acc += q[d] * k[d];
      v += acc;
    }
    if (pmask[n * LL_ + l]) v = -INFINITY;
    sc[l] = v;
  }
  for (int m = 0; m < PL_; m++) {
    const float* k = kvp + ((size_t)b * PL_ + m) * (2 * C_) + h * HD_;
    float acc = 0.f;
#pragma unroll
    for (int d = 0; d < HD_; d++) acc += q[d] * k[d];
    sc[LL_ + m] = acc + cpb[(size_t)rpi[n * PL_ + m] * NH_ + h];
  }
  float mx = -INFINITY;
  for (int i = 0; i < LL_ + PL_; i++) mx = fmaxf(mx, sc[i]);
  float sum = 0.f;
  for (int i = 0; i < LL_ + PL_; i++) {
    float e = expf(sc[i] - mx);
    sc[i] = e; sum += e;
  }
  float rs = 1.0f / sum;
  for (int i = 0; i < LL_ + PL_; i++) sc[i] *= rs;

  float o[HD_];
#pragma unroll
  for (int d = 0; d < HD_; d++) o[d] = 0.f;

#pragma unroll
  for (int l = 0; l < LL_; l++) {
    float ql = 0.f;
#pragma unroll
    for (int d = 0; d < HD_; d++) {
      float qn = q[d] * invs - qe[h * HD_ + d];
      ql += qn * lt[(h * HD_ + d) * LL_ + l];
    }
    float coef = ql + lb[h * LL_ + l] + sc[l];
    if (ok[l]) {
      const float* v = kv + ((size_t)b * N_ + nbr[l]) * (2 * C_) + C_ + h * HD_;
#pragma unroll
      for (int d = 0; d < HD_; d++) o[d] += coef * v[d];
    }
  }
  for (int m = 0; m < PL_; m++) {
    float w = sc[LL_ + m];
    const float* v = kvp + ((size_t)b * PL_ + m) * (2 * C_) + C_ + h * HD_;
#pragma unroll
    for (int d = 0; d < HD_; d++) o[d] += w * v[d];
  }
  __bf16* orow = obuf + ((size_t)b * N_ + n) * C_ + h * HD_;
#pragma unroll
  for (int d = 0; d < HD_; d++) orow[d] = (__bf16)o[d];
}

// depthwise 3x3 on u + GELU, times v:  uv bf16 [M,2048] -> m bf16 [M,1024]
__global__ void dwglu_kernel(const __bf16* __restrict__ uv,
                             const float* __restrict__ dw,
                             const float* __restrict__ dwb,
                             __bf16* __restrict__ mout) {
  int idx = blockIdx.x * blockDim.x + threadIdx.x;
  if (idx >= B_ * N_ * HID_) return;
  int b = idx / (N_ * HID_);
  int rem = idx % (N_ * HID_);
  int n = rem / HID_, ch = rem % HID_;
  int y0 = n / W_, x0 = n % W_;
  float acc = dwb[ch];
#pragma unroll
  for (int ky = 0; ky < 3; ky++) {
    int yy = y0 + ky - 1;
    if (yy < 0 || yy >= H_) continue;
#pragma unroll
    for (int kx = 0; kx < 3; kx++) {
      int xx = x0 + kx - 1;
      if (xx < 0 || xx >= W_) continue;
      float u = (float)uv[((size_t)b * N_ + yy * W_ + xx) * (2 * HID_) + ch];
      acc += u * dw[ch * 9 + ky * 3 + kx];
    }
  }
  float v = (float)uv[((size_t)b * N_ + n) * (2 * HID_) + HID_ + ch];
  mout[((size_t)b * N_ + n) * HID_ + ch] = (__bf16)(gelu_exact(acc) * v);
}

// ---------------------------------------------------------------------------
extern "C" void kernel_launch(void* const* d_in, const int* in_sizes, int n_in,
                              void* d_out, int out_size, void* d_ws,
                              size_t ws_size, hipStream_t stream) {
  const float* x    = (const float*)d_in[0];
  const int*   rpi  = (const int*)d_in[3];
  const float* rct  = (const float*)d_in[4];
  const float* sls  = (const float*)d_in[5];
  const unsigned char* pmask = (const unsigned char*)d_in[6];
  const float* n1g  = (const float*)d_in[7];
  const float* n1b  = (const float*)d_in[8];
  const float* q_w  = (const float*)d_in[9];
  const float* q_b  = (const float*)d_in[10];
  const float* kv_w = (const float*)d_in[11];
  const float* kv_b = (const float*)d_in[12];
  const float* temp = (const float*)d_in[13];
  const float* qe   = (const float*)d_in[14];
  const float* rpb  = (const float*)d_in[15];
  const float* lt   = (const float*)d_in[16];
  const float* lb   = (const float*)d_in[17];
  const float* sr_w = (const float*)d_in[18];
  const float* sr_b = (const float*)d_in[19];
  const float* nsg  = (const float*)d_in[20];
  const float* nsb  = (const float*)d_in[21];
  const float* c1w  = (const float*)d_in[22];
  const float* c1b  = (const float*)d_in[23];
  const float* c2w  = (const float*)d_in[24];
  const float* c2b  = (const float*)d_in[25];
  const float* pj_w = (const float*)d_in[26];
  const float* pj_b = (const float*)d_in[27];
  const float* n2g  = (const float*)d_in[28];
  const float* n2b  = (const float*)d_in[29];
  const float* f1w  = (const float*)d_in[30];
  const float* f1b  = (const float*)d_in[31];
  const float* dww  = (const float*)d_in[32];
  const float* dwb  = (const float*)d_in[33];
  const float* f2w  = (const float*)d_in[34];
  const float* f2b  = (const float*)d_in[35];

  const int T = in_sizes[4] / 2;  // (H+pH-1)*(W+pW-1)
  const int M = M_;
  char* ws = (char*)d_ws;
  auto al = [](size_t v) { return (v + 255) & ~(size_t)255; };

  size_t offA = 0;
  size_t szA = al((size_t)M * C_ * sizeof(__bf16));  // h / o / y (bf16)
  size_t off1 = offA + szA;                          // q+s f32, later uv bf16
  size_t sz1 = al((size_t)2 * M * C_ * 4 > (size_t)M * 2 * HID_ * 2
                      ? (size_t)2 * M * C_ * 4
                      : (size_t)M * 2 * HID_ * 2);
  size_t off2 = off1 + sz1;                          // kv f32, later m bf16
  size_t sz2 = al((size_t)M * 2 * C_ * 4);
  size_t off3 = off2 + sz2;                          // x2 f32
  size_t sz3 = al((size_t)M * C_ * 4);
  size_t off4 = off3 + sz3;
  size_t offPool = off4;
  size_t offSln  = offPool + al((size_t)B_ * PL_ * C_ * 4);
  size_t offKvp  = offSln + al((size_t)B_ * PL_ * C_ * 2);
  size_t offCpb  = offKvp + al((size_t)B_ * PL_ * 2 * C_ * 4);

  __bf16* hbuf   = (__bf16*)(ws + offA);
  float*  qbuf   = (float*)(ws + off1);
  float*  sbuf   = (float*)(ws + off1 + (size_t)M * C_ * 4);
  __bf16* uvbuf  = (__bf16*)(ws + off1);
  float*  kvbuf  = (float*)(ws + off2);
  __bf16* mbuf   = (__bf16*)(ws + off2);
  float*  x2buf  = (float*)(ws + off3);
  float*  poolb  = (float*)(ws + offPool);
  __bf16* slnbuf = (__bf16*)(ws + offSln);
  float*  kvpbuf = (float*)(ws + offKvp);
  float*  cpbbuf = (float*)(ws + offCpb);

  dim3 blk256(256), blk128(128);
  int rowBlocks = (M + 127) / 128;

  // 1) LN1 -> h (bf16)
  ln_bf16_kernel<<<M, blk128, 0, stream>>>(x, n1g, n1b, hbuf, M, C_);

  // 2) q / kv / sr GEMMs (WMMA bf16)
  gemm_bf16_kernel<<<dim3(C_ / 64, rowBlocks), blk256, 0, stream>>>(
      hbuf, q_w, q_b, nullptr, qbuf, nullptr, M, C_, C_, 0);
  gemm_bf16_kernel<<<dim3(2 * C_ / 64, rowBlocks), blk256, 0, stream>>>(
      hbuf, kv_w, kv_b, nullptr, kvbuf, nullptr, M, 2 * C_, C_, 0);
  gemm_bf16_kernel<<<dim3(C_ / 64, rowBlocks), blk256, 0, stream>>>(
      hbuf, sr_w, sr_b, nullptr, sbuf, nullptr, M, C_, C_, 1 /*gelu*/);

  // 3) q post + k l2-norm (in place)
  q_post_kernel<<<(M * NH_ + 255) / 256, blk256, 0, stream>>>(qbuf, temp, qe,
                                                              sls);
  k_norm_kernel<<<(M * NH_ + 255) / 256, blk256, 0, stream>>>(kvbuf, M);

  // 4) pool -> LN -> bf16 -> pooled kv GEMM -> k_p l2-norm
  pool_kernel<<<(B_ * PL_ * C_ + 255) / 256, blk256, 0, stream>>>(sbuf, poolb);
  ln_bf16_kernel<<<B_ * PL_, blk128, 0, stream>>>(poolb, nsg, nsb, slnbuf,
                                                  B_ * PL_, C_);
  gemm_bf16_kernel<<<dim3(2 * C_ / 64, (B_ * PL_ + 127) / 128), blk256, 0,
                    stream>>>(slnbuf, kv_w, kv_b, nullptr, kvpbuf, nullptr,
                              B_ * PL_, 2 * C_, C_, 0);
  k_norm_kernel<<<(B_ * PL_ * NH_ + 255) / 256, blk256, 0, stream>>>(kvpbuf,
                                                                     B_ * PL_);

  // 5) CPB MLP
  cpb_kernel<<<(T + 255) / 256, blk256, 0, stream>>>(rct, c1w, c1b, c2w, c2b,
                                                     cpbbuf, T);

  // 6) attention core -> o (bf16, reuses region A)
  attn_kernel<<<(B_ * NH_ * N_ + 255) / 256, blk256, 0, stream>>>(
      qbuf, kvbuf, kvpbuf, cpbbuf, rpi, pmask, rpb, lt, lb, temp, sls, qe,
      hbuf);

  // 7) proj GEMM with residual x -> x2 ; LN2 -> y (bf16, region A again)
  gemm_bf16_kernel<<<dim3(C_ / 64, rowBlocks), blk256, 0, stream>>>(
      hbuf, pj_w, pj_b, x, x2buf, nullptr, M, C_, C_, 0);
  ln_bf16_kernel<<<M, blk128, 0, stream>>>(x2buf, n2g, n2b, hbuf, M, C_);

  // 8) fc1 -> uv (bf16) ; dwconv+GLU -> m (bf16) ; fc2 + residual -> out
  gemm_bf16_kernel<<<dim3(2 * HID_ / 64, rowBlocks), blk256, 0, stream>>>(
      hbuf, f1w, f1b, nullptr, nullptr, uvbuf, M, 2 * HID_, C_, 0);
  dwglu_kernel<<<(B_ * N_ * HID_ + 255) / 256, blk256, 0, stream>>>(
      uvbuf, dww, dwb, mbuf);
  gemm_bf16_kernel<<<dim3(C_ / 64, rowBlocks), blk256, 0, stream>>>(
      mbuf, f2w, f2b, x2buf, (float*)d_out, nullptr, M, C_, HID_, 0);

  (void)n_in; (void)out_size; (void)ws_size; (void)in_sizes;
}
